// RMHSA_48060684042761
// MI455X (gfx1250) — matrix operand order repaired and checked by reading, not compiled, not run on hardware
//
#include <hip/hip_runtime.h>
#include <hip/hip_bf16.h>

#define D_MODEL 1024
#define NUM_HEADS 16
#define D_HEAD 64
#define SEQ 1024
#define BATCH 2
#define NROWS (BATCH * SEQ)      // 2048
#define ER_ROWS_PAD 2112         // 2047 valid + zero pad (band reads up to row 1055)

typedef __bf16 bf16_t;
typedef __attribute__((ext_vector_type(16))) __bf16 v16bf;
typedef __attribute__((ext_vector_type(8)))  float  v8f;
typedef unsigned int u32x4 __attribute__((ext_vector_type(4)));
typedef int i32x8 __attribute__((ext_vector_type(8)));
typedef int i32x4 __attribute__((ext_vector_type(4)));

#if __has_builtin(__builtin_amdgcn_tensor_load_to_lds) && \
    __has_builtin(__builtin_amdgcn_s_wait_tensorcnt)
#define USE_TDM 1
#endif

union FragU {
    v16bf v;
    uint4 q[2];
};

// A-matrix 16x32 bf16 fragment (row-major source, leading dim ld in elements).
// lane<16: row=lane, K=[0..7],[16..23]; lane>=16: row=lane-16, K=[8..15],[24..31]
__device__ __forceinline__ v16bf load_fragA(const bf16_t* p, int ld, int lane) {
    int m = lane & 15, half = lane >> 4;
    const bf16_t* r = p + (size_t)m * ld + half * 8;
    FragU u;
    u.q[0] = *reinterpret_cast<const uint4*>(r);
    u.q[1] = *reinterpret_cast<const uint4*>(r + 16);
    return u.v;
}

// B-matrix 32x16 bf16 fragment from B^T stored row-major [N][K] (ld in elements).
// lane<16: col=lane, K=[0..7],[16..23]; lane>=16: col=lane-16, K=[8..15],[24..31]
__device__ __forceinline__ v16bf load_fragB(const bf16_t* p, int ld, int lane) {
    int n = lane & 15, half = lane >> 4;
    const bf16_t* r = p + (size_t)n * ld + half * 8;
    FragU u;
    u.q[0] = *reinterpret_cast<const uint4*>(r);
    u.q[1] = *reinterpret_cast<const uint4*>(r + 16);
    return u.v;
}

#define WMMA_BF16(a, b, c) \
    __builtin_amdgcn_wmma_f32_16x16x32_bf16(false, (a), false, (b), (short)0, (c), false, false)

#ifdef USE_TDM
// TDM: async DMA of a 64-row x KCHUNK-col bf16 tile of Wt (row stride 1024 elems)
// into LDS at byte offset lds_off (packed rows of KCHUNK elems). D# per CDNA5 ISA §8.
// This toolchain's builtin takes 6 args: (g0, g1, g2, g3, extra, cpol).
__device__ __forceinline__ void tdm_load_b_tile(const bf16_t* gsrc, unsigned lds_off,
                                                int kchunk) {
    unsigned long long ga = (unsigned long long)(uintptr_t)gsrc;
    u32x4 g0;
    g0[0] = 1u;                                   // count=1, user mode, no gather
    g0[1] = lds_off;                              // lds_addr (bytes)
    g0[2] = (unsigned)ga;                         // global_addr[31:0]
    g0[3] = (unsigned)((ga >> 32) & 0x01FFFFFFu)  // global_addr[56:32]
            | 0x80000000u;                        // type=2 ("image") at bits 127:126
    i32x8 g1;
    g1[0] = 0x00010000;                           // workgroup_mask=0, data_size=1 (2B)
    g1[1] = (int)(1024u << 16);                   // tensor_dim0 low16 = 1024
    g1[2] = (int)(1024u << 16);                   // tensor_dim0 hi=0 | tensor_dim1 lo16=1024
    g1[3] = (int)((unsigned)kchunk << 16);        // tensor_dim1 hi=0 | tile_dim0=kchunk
    g1[4] = 64;                                   // tile_dim1=64, tile_dim2=0 (2D)
    g1[5] = 1024;                                 // tensor_dim0_stride low32 = 1024
    g1[6] = 0;                                    // stride0 hi | tensor_dim1_stride lo (unused)
    g1[7] = 0;
    i32x4 z4 = {0, 0, 0, 0};                      // groups 2/3 zero (tile_dim3=0 => unused)
    i32x8 z8 = {0, 0, 0, 0, 0, 0, 0, 0};
    __builtin_amdgcn_tensor_load_to_lds(g0, g1, z4, z4, z8, 0);
}
#endif

// ---------------- prep kernels ----------------

__global__ void k_cvt_x(const float* __restrict__ x, bf16_t* __restrict__ xb, int n) {
    int i = blockIdx.x * blockDim.x + threadIdx.x;
    if (i < n) xb[i] = (bf16_t)x[i];
}

__global__ void k_cvt_er(const float* __restrict__ er, bf16_t* __restrict__ erb) {
    int i = blockIdx.x * blockDim.x + threadIdx.x;
    if (i >= ER_ROWS_PAD * D_HEAD) return;
    erb[i] = (i < (2 * SEQ - 1) * D_HEAD) ? (bf16_t)er[i] : (bf16_t)0.0f;
}

// W (KxN fp32, row-major) -> Wt (NxK bf16, row-major) via LDS-tiled transpose
__global__ __launch_bounds__(256) void k_transpose_w(
    const float* __restrict__ wq, const float* __restrict__ wk,
    const float* __restrict__ wv, const float* __restrict__ wo,
    bf16_t* __restrict__ wqt, bf16_t* __restrict__ wkt,
    bf16_t* __restrict__ wvt, bf16_t* __restrict__ wot) {
    __shared__ float tile[32][33];
    int z = blockIdx.z;
    const float* w = (z == 0) ? wq : (z == 1) ? wk : (z == 2) ? wv : wo;
    bf16_t* wt = (z == 0) ? wqt : (z == 1) ? wkt : (z == 2) ? wvt : wot;
    int tx = threadIdx.x & 31;
    int ty = threadIdx.x >> 5;
    int kbase = blockIdx.y * 32;
    int nbase = blockIdx.x * 32;
#pragma unroll
    for (int i = 0; i < 32; i += 8)
        tile[ty + i][tx] = w[(size_t)(kbase + ty + i) * D_MODEL + nbase + tx];
    __syncthreads();
#pragma unroll
    for (int i = 0; i < 32; i += 8)
        wt[(size_t)(nbase + ty + i) * D_MODEL + kbase + tx] = (bf16_t)tile[tx][ty + i];
}

// ---------------- GEMM core ----------------
// 128 threads = 4 waves; workgroup computes 128 rows x 64 cols; wave w owns 32 rows.
// B tile (64 cols x 128 k, bf16) staged in LDS via TDM (double buffered) or
// cooperative copy fallback. A fragments loaded directly from global (L2 resident).

#define KCHUNK 128
#define NKOUT (D_MODEL / KCHUNK)   // 8

struct GemmRegs {
    v8f acc[2][4];
};

__device__ __forceinline__ void gemm_compute_chunk(
    GemmRegs& R, const bf16_t* abase, int lda, const bf16_t* bbuf, int lane) {
#pragma unroll
    for (int kk = 0; kk < KCHUNK; kk += 32) {
        v16bf a0 = load_fragA(abase + kk, lda, lane);
        v16bf a1 = load_fragA(abase + (size_t)16 * lda + kk, lda, lane);
        v16bf b0 = load_fragB(bbuf + 0 * 16 * KCHUNK + kk, KCHUNK, lane);
        v16bf b1 = load_fragB(bbuf + 1 * 16 * KCHUNK + kk, KCHUNK, lane);
        v16bf b2 = load_fragB(bbuf + 2 * 16 * KCHUNK + kk, KCHUNK, lane);
        v16bf b3 = load_fragB(bbuf + 3 * 16 * KCHUNK + kk, KCHUNK, lane);
        R.acc[0][0] = WMMA_BF16(a0, b0, R.acc[0][0]);
        R.acc[0][1] = WMMA_BF16(a0, b1, R.acc[0][1]);
        R.acc[0][2] = WMMA_BF16(a0, b2, R.acc[0][2]);
        R.acc[0][3] = WMMA_BF16(a0, b3, R.acc[0][3]);
        R.acc[1][0] = WMMA_BF16(a1, b0, R.acc[1][0]);
        R.acc[1][1] = WMMA_BF16(a1, b1, R.acc[1][1]);
        R.acc[1][2] = WMMA_BF16(a1, b2, R.acc[1][2]);
        R.acc[1][3] = WMMA_BF16(a1, b3, R.acc[1][3]);
    }
}

// Runs the whole K loop for one (r0, c0) wave tile. bstage: [2][64][KCHUNK] bf16 in LDS.
__device__ __forceinline__ void gemm_mainloop(
    GemmRegs& R, const bf16_t* a_rows, int lda, const bf16_t* wt, int c0,
    bf16_t (*bstage)[64][KCHUNK], int wave, int lane) {
#pragma unroll
    for (int rt = 0; rt < 2; ++rt)
#pragma unroll
        for (int nt = 0; nt < 4; ++nt) R.acc[rt][nt] = v8f{};

#ifdef USE_TDM
    if (wave == 0)
        tdm_load_b_tile(wt + (size_t)c0 * D_MODEL, (unsigned)(uintptr_t)&bstage[0][0][0],
                        KCHUNK);
    for (int i = 0; i < NKOUT - 1; ++i) {
        if (wave == 0)
            tdm_load_b_tile(wt + (size_t)c0 * D_MODEL + (i + 1) * KCHUNK,
                            (unsigned)(uintptr_t)&bstage[(i + 1) & 1][0][0], KCHUNK);
        __builtin_amdgcn_s_wait_tensorcnt((short)1);   // buffer i is complete
        __syncthreads();
        gemm_compute_chunk(R, a_rows + i * KCHUNK, lda, &bstage[i & 1][0][0], lane);
        __syncthreads();                               // done reading before re-fill
    }
    __builtin_amdgcn_s_wait_tensorcnt((short)0);
    __syncthreads();
    gemm_compute_chunk(R, a_rows + (NKOUT - 1) * KCHUNK, lda,
                       &bstage[(NKOUT - 1) & 1][0][0], lane);
#else
    int tid = wave * 32 + lane;
    for (int i = 0; i < NKOUT; ++i) {
        // cooperative copy: 64 rows x KCHUNK elems = 64*KCHUNK*2 bytes, 16B chunks
        const int nchunks = 64 * KCHUNK / 8;
        for (int c = tid; c < nchunks; c += 128) {
            int n = c / (KCHUNK / 8);
            int q = c % (KCHUNK / 8);
            *reinterpret_cast<uint4*>(&bstage[0][n][q * 8]) =
                *reinterpret_cast<const uint4*>(wt + (size_t)(c0 + n) * D_MODEL +
                                                i * KCHUNK + q * 8);
        }
        __syncthreads();
        gemm_compute_chunk(R, a_rows + i * KCHUNK, lda, &bstage[0][0][0], lane);
        __syncthreads();
    }
#endif
}

// ---------------- fused QKV projection GEMM ----------------
__global__ __launch_bounds__(128) void k_qkv_gemm(
    const bf16_t* __restrict__ xb,
    const bf16_t* __restrict__ wqt, const bf16_t* __restrict__ wkt, const bf16_t* __restrict__ wvt,
    const float* __restrict__ bq, const float* __restrict__ bk, const float* __restrict__ bv,
    bf16_t* __restrict__ qh, bf16_t* __restrict__ kh, bf16_t* __restrict__ vtr) {
    __shared__ __align__(128) bf16_t bstage[2][64][KCHUNK];
    int lane = threadIdx.x & 31;
    int wave = threadIdx.x >> 5;
    int r0 = blockIdx.x * 128 + wave * 32;
    int c0 = blockIdx.y * 64;
    int z = blockIdx.z;
    const bf16_t* wt = (z == 0) ? wqt : (z == 1) ? wkt : wvt;
    const float* bias = (z == 0) ? bq : (z == 1) ? bk : bv;

    GemmRegs R;
    gemm_mainloop(R, xb + (size_t)r0 * D_MODEL, D_MODEL, wt, c0, bstage, wave, lane);

    int nlane = lane & 15;
    int rowoff = 8 * (lane >> 4);
#pragma unroll
    for (int rt = 0; rt < 2; ++rt) {
#pragma unroll
        for (int nt = 0; nt < 4; ++nt) {
            int c = c0 + nt * 16 + nlane;
            float bval = bias[c];
            int h = c >> 6, d = c & 63;
#pragma unroll
            for (int e = 0; e < 8; ++e) {
                int gr = r0 + rt * 16 + rowoff + e;
                int bb = gr >> 10, s = gr & 1023;
                bf16_t val = (bf16_t)(R.acc[rt][nt][e] + bval);
                size_t hb = (size_t)(bb * NUM_HEADS + h);
                if (z == 0)      qh[(hb * SEQ + s) * D_HEAD + d] = val;
                else if (z == 1) kh[(hb * SEQ + s) * D_HEAD + d] = val;
                else             vtr[(hb * D_HEAD + d) * SEQ + s] = val;  // V transposed
            }
        }
    }
}

// ---------------- output projection GEMM (fp32 out + bias) ----------------
__global__ __launch_bounds__(128) void k_out_gemm(
    const bf16_t* __restrict__ attn, const bf16_t* __restrict__ wot,
    const float* __restrict__ bo, float* __restrict__ out) {
    __shared__ __align__(128) bf16_t bstage[2][64][KCHUNK];
    int lane = threadIdx.x & 31;
    int wave = threadIdx.x >> 5;
    int r0 = blockIdx.x * 128 + wave * 32;
    int c0 = blockIdx.y * 64;

    GemmRegs R;
    gemm_mainloop(R, attn + (size_t)r0 * D_MODEL, D_MODEL, wot, c0, bstage, wave, lane);

    int nlane = lane & 15;
    int rowoff = 8 * (lane >> 4);
#pragma unroll
    for (int rt = 0; rt < 2; ++rt) {
#pragma unroll
        for (int nt = 0; nt < 4; ++nt) {
            int c = c0 + nt * 16 + nlane;
            float bval = bo[c];
#pragma unroll
            for (int e = 0; e < 8; ++e) {
                int gr = r0 + rt * 16 + rowoff + e;
                out[(size_t)gr * D_MODEL + c] = R.acc[rt][nt][e] + bval;
            }
        }
    }
}

// ---------------- flash attention with relative position (skew) ----------------
// grid (B*H, S/64), block 128: wave w handles a 16-row q-tile; waves independent.
__global__ __launch_bounds__(128) void k_attn(
    const bf16_t* __restrict__ qh, const bf16_t* __restrict__ kh,
    const bf16_t* __restrict__ vtr, const bf16_t* __restrict__ erb,
    bf16_t* __restrict__ attn) {
    __shared__ float skew[4][16][49];   // per-wave 16x48 QEr band (+pad)
    __shared__ bf16_t pl[4][16][32];    // per-wave P tile

    int lane = threadIdx.x & 31;
    int wave = threadIdx.x >> 5;
    int bh = blockIdx.x;
    int q0 = blockIdx.y * 64 + wave * 16;

    const bf16_t* Qp = qh + (size_t)bh * SEQ * D_HEAD;
    const bf16_t* Kp = kh + (size_t)bh * SEQ * D_HEAD;
    const bf16_t* Vp = vtr + (size_t)bh * D_HEAD * SEQ;

    v16bf qa0 = load_fragA(Qp + (size_t)q0 * D_HEAD + 0, D_HEAD, lane);
    v16bf qa1 = load_fragA(Qp + (size_t)q0 * D_HEAD + 32, D_HEAD, lane);

    v8f o[4] = {};
    float mrow[8], lrow[8];
#pragma unroll
    for (int e = 0; e < 8; ++e) { mrow[e] = -3.0e38f; lrow[e] = 0.0f; }

    float* skewp = &skew[wave][0][0];
    bf16_t* plp = &pl[wave][0][0];
    int nlane = lane & 15;
    int rowoff = 8 * (lane >> 4);
    const float scale = 0.125f;  // 1/sqrt(64)

    int kend = q0 + 16;
    for (int k0 = 0; k0 < kend; k0 += 32) {
        // scores = Q @ K^T
        v8f sc[2] = {};
#pragma unroll
        for (int t = 0; t < 2; ++t) {
            const bf16_t* kb = Kp + (size_t)(k0 + t * 16) * D_HEAD;
            v16bf b0 = load_fragB(kb + 0, D_HEAD, lane);
            v16bf b1 = load_fragB(kb + 32, D_HEAD, lane);
            sc[t] = WMMA_BF16(qa0, b0, sc[t]);
            sc[t] = WMMA_BF16(qa1, b1, sc[t]);
        }
        // QEr band: T[qi][c] = q . Er[off+c], c in [0,48); Srel[qi][kj]=T[qi][kj-qi+15]
        int off = (SEQ - 1) + k0 - q0 - 15;
        const bf16_t* eb = erb + (size_t)off * D_HEAD;
#pragma unroll
        for (int j = 0; j < 3; ++j) {
            v8f tf = {};
            const bf16_t* ej = eb + (size_t)j * 16 * D_HEAD;
            v16bf b0 = load_fragB(ej + 0, D_HEAD, lane);
            v16bf b1 = load_fragB(ej + 32, D_HEAD, lane);
            tf = WMMA_BF16(qa0, b0, tf);
            tf = WMMA_BF16(qa1, b1, tf);
#pragma unroll
            for (int e = 0; e < 8; ++e)
                skewp[(rowoff + e) * 49 + j * 16 + nlane] = tf[e];
        }
        __asm__ volatile("s_wait_dscnt 0" ::: "memory");

        // add skewed Srel, scale, causal mask; iteration row max
        float itmax[8];
#pragma unroll
        for (int t = 0; t < 2; ++t) {
#pragma unroll
            for (int e = 0; e < 8; ++e) {
                int qi = rowoff + e;
                int kj = t * 16 + nlane;
                float s = (sc[t][e] + skewp[qi * 49 + (kj - qi + 15)]) * scale;
                if (k0 + kj > q0 + qi) s = -3.0e38f;
                sc[t][e] = s;
                itmax[e] = (t == 0) ? s : fmaxf(itmax[e], s);
            }
        }
#pragma unroll
        for (int m = 1; m < 16; m <<= 1)
#pragma unroll
            for (int e = 0; e < 8; ++e)
                itmax[e] = fmaxf(itmax[e], __shfl_xor(itmax[e], m, 32));

        // online softmax
        float psum[8];
#pragma unroll
        for (int e = 0; e < 8; ++e) {
            float mnew = fmaxf(mrow[e], itmax[e]);
            float esc = __expf(mrow[e] - mnew);
            mrow[e] = mnew;
            lrow[e] *= esc;
#pragma unroll
            for (int j2 = 0; j2 < 4; ++j2) o[j2][e] = o[j2][e] * esc;
            float p0 = __expf(sc[0][e] - mnew);
            float p1 = __expf(sc[1][e] - mnew);
            sc[0][e] = p0; sc[1][e] = p1;
            psum[e] = p0 + p1;
        }
#pragma unroll
        for (int m = 1; m < 16; m <<= 1)
#pragma unroll
            for (int e = 0; e < 8; ++e)
                psum[e] += __shfl_xor(psum[e], m, 32);
#pragma unroll
        for (int e = 0; e < 8; ++e) lrow[e] += psum[e];

        // P tile -> bf16 LDS -> A fragment; O += P @ V
#pragma unroll
        for (int t = 0; t < 2; ++t)
#pragma unroll
            for (int e = 0; e < 8; ++e)
                plp[(rowoff + e) * 32 + t * 16 + nlane] = (bf16_t)sc[t][e];
        __asm__ volatile("s_wait_dscnt 0" ::: "memory");
        v16bf pa = load_fragA(plp, 32, lane);
#pragma unroll
        for (int j2 = 0; j2 < 4; ++j2) {
            v16bf vb = load_fragB(Vp + (size_t)(j2 * 16) * SEQ + k0, SEQ, lane);
            o[j2] = WMMA_BF16(pa, vb, o[j2]);
        }
    }

    // normalize + store concat layout attn[b][s][h*64+d]
#pragma unroll
    for (int e = 0; e < 8; ++e) lrow[e] = 1.0f / lrow[e];
    int bb = bh >> 4, h = bh & 15;
#pragma unroll
    for (int j2 = 0; j2 < 4; ++j2) {
        int d = j2 * 16 + nlane;
#pragma unroll
        for (int e = 0; e < 8; ++e) {
            int s = q0 + rowoff + e;
            attn[((size_t)(bb * SEQ + s)) * D_MODEL + h * D_HEAD + d] =
                (bf16_t)(o[j2][e] * lrow[e]);
        }
    }
}

// ---------------- host launcher ----------------

extern "C" void kernel_launch(void* const* d_in, const int* in_sizes, int n_in,
                              void* d_out, int out_size, void* d_ws, size_t ws_size,
                              hipStream_t stream) {
    const float* x  = (const float*)d_in[0];
    const float* Wq = (const float*)d_in[1];
    const float* bq = (const float*)d_in[2];
    const float* Wk = (const float*)d_in[3];
    const float* bk = (const float*)d_in[4];
    const float* Wv = (const float*)d_in[5];
    const float* bv = (const float*)d_in[6];
    const float* Wo = (const float*)d_in[7];
    const float* bo = (const float*)d_in[8];
    const float* Er = (const float*)d_in[9];
    float* out = (float*)d_out;

    char* ws = (char*)d_ws;
    size_t off = 0;
    auto alloc = [&](size_t bytes) -> void* {
        void* p = ws + off;
        off += (bytes + 255) & ~(size_t)255;
        return p;
    };
    bf16_t* xb   = (bf16_t*)alloc((size_t)NROWS * D_MODEL * 2);
    bf16_t* wqt  = (bf16_t*)alloc((size_t)D_MODEL * D_MODEL * 2);
    bf16_t* wkt  = (bf16_t*)alloc((size_t)D_MODEL * D_MODEL * 2);
    bf16_t* wvt  = (bf16_t*)alloc((size_t)D_MODEL * D_MODEL * 2);
    bf16_t* wot  = (bf16_t*)alloc((size_t)D_MODEL * D_MODEL * 2);
    bf16_t* erb  = (bf16_t*)alloc((size_t)ER_ROWS_PAD * D_HEAD * 2);
    bf16_t* qhp  = (bf16_t*)alloc((size_t)BATCH * NUM_HEADS * SEQ * D_HEAD * 2);
    bf16_t* khp  = (bf16_t*)alloc((size_t)BATCH * NUM_HEADS * SEQ * D_HEAD * 2);
    bf16_t* vtp  = (bf16_t*)alloc((size_t)BATCH * NUM_HEADS * D_HEAD * SEQ * 2);
    bf16_t* attn = (bf16_t*)alloc((size_t)NROWS * D_MODEL * 2);
    (void)ws_size; (void)in_sizes; (void)n_in; (void)out_size;

    int nx = NROWS * D_MODEL;
    k_cvt_x<<<dim3((nx + 255) / 256), dim3(256), 0, stream>>>(x, xb, nx);
    k_cvt_er<<<dim3((ER_ROWS_PAD * D_HEAD + 255) / 256), dim3(256), 0, stream>>>(Er, erb);
    k_transpose_w<<<dim3(32, 32, 4), dim3(256), 0, stream>>>(Wq, Wk, Wv, Wo, wqt, wkt, wvt, wot);
    k_qkv_gemm<<<dim3(NROWS / 128, D_MODEL / 64, 3), dim3(128), 0, stream>>>(
        xb, wqt, wkt, wvt, bq, bk, bv, qhp, khp, vtp);
    k_attn<<<dim3(BATCH * NUM_HEADS, SEQ / 64), dim3(128), 0, stream>>>(
        qhp, khp, vtp, erb, attn);
    k_out_gemm<<<dim3(NROWS / 128, D_MODEL / 64), dim3(128), 0, stream>>>(attn, wot, bo, out);
}
